// MambaFFNBlock_1889785610991
// MI455X (gfx1250) — matrix operand order repaired and checked by reading, not compiled, not run on hardware
//
#include <hip/hip_runtime.h>

// ---------------- problem constants ----------------
#define B_       2
#define T_       2048
#define DMODEL   1024
#define DINNER   2048
#define DSTATE   128
#define DCONV    4
#define HEADDIM  64
#define NHEADS   32
#define CONVDIM  2304      // DINNER + 2*DSTATE
#define DINPROJ  4384      // 2*DINNER + 2*DSTATE + NHEADS
#define ROWS     (B_ * T_) // 4096

// ---------------- CDNA5 vector types ----------------
typedef __attribute__((ext_vector_type(16))) __bf16 v16bf;
typedef __attribute__((ext_vector_type(8)))  __bf16 v8bf;
typedef __attribute__((ext_vector_type(8)))  float  v8f;

static __device__ __forceinline__ unsigned short f32_to_bf16(float f) {
  unsigned int u = __builtin_bit_cast(unsigned int, f);
  u += 0x7FFFu + ((u >> 16) & 1u);   // round-to-nearest-even
  return (unsigned short)(u >> 16);
}

static __device__ __forceinline__ float sigmoidf_(float x) {
  return 1.f / (1.f + __expf(-x));
}

// 256-thread block reduction (wave32: shfl width 32, 8 waves)
static __device__ __forceinline__ float block_sum_256(float v, float* s8) {
  #pragma unroll
  for (int off = 16; off > 0; off >>= 1) v += __shfl_xor(v, off, 32);
  const int tid = threadIdx.x;
  if ((tid & 31) == 0) s8[tid >> 5] = v;
  __syncthreads();
  float r = 0.f;
  #pragma unroll
  for (int i = 0; i < 8; ++i) r += s8[i];
  __syncthreads();
  return r;
}

// ---------------- K0: RMSNorm(x) -> bf16 ----------------
__global__ __launch_bounds__(256)
void rmsnorm_pack_kernel(const float* __restrict__ x, const float* __restrict__ w,
                         unsigned short* __restrict__ xn) {
  __shared__ float s8[8];
  const int row = blockIdx.x;            // 0..4095
  const int tid = threadIdx.x;
  const float* xr = x + (size_t)row * DMODEL;
  float ss = 0.f;
  #pragma unroll
  for (int j = 0; j < 4; ++j) { float v = xr[tid + j * 256]; ss += v * v; }
  const float tot = block_sum_256(ss, s8);
  const float scale = rsqrtf(tot / (float)DMODEL + 1e-5f);
  #pragma unroll
  for (int j = 0; j < 4; ++j) {
    const int i = tid + j * 256;
    xn[(size_t)row * DMODEL + i] = f32_to_bf16(xr[i] * scale * w[i]);
  }
}

// ---------------- K1: W[K x N] -> Wt[N x K] bf16 ----------------
__global__ __launch_bounds__(256)
void transpose_pack_kernel(const float* __restrict__ W, unsigned short* __restrict__ Wt,
                           int K, int N) {
  const size_t i = (size_t)blockIdx.x * 256 + threadIdx.x;
  if (i >= (size_t)K * N) return;
  const int n = (int)(i / K);
  const int k = (int)(i % K);
  Wt[i] = f32_to_bf16(W[(size_t)k * N + n]);
}

// ---------------- K2: bf16 WMMA GEMM  O = A(MxK) * Bt(NxK)^T (+Res) ----------
// Block tile 128x128, 256 threads = 8 waves in 4(M) x 2(N); each wave: 2x4
// 16x16 WMMA tiles; K-step 32.  LDS rows padded to 40 halves (80B) so the
// 16-row fragment gathers hit 16 distinct banks.
// Global->LDS staging uses gfx1250 GLOBAL_LOAD_ASYNC_TO_LDS_B128 (ASYNCcnt),
// double-buffered: async-load tile k+1 while WMMAs consume tile k.
#define TILE_M 128
#define TILE_N 128
#define TILE_K 32
#define LDSROW 40
#define BUFHALVES (TILE_M * LDSROW + TILE_N * LDSROW)   // 10240 halves / 20480 B

static __device__ __forceinline__ v16bf cat8(v8bf lo, v8bf hi) {
  return __builtin_shufflevector(lo, hi, 0,1,2,3,4,5,6,7,8,9,10,11,12,13,14,15);
}

// LDS byte offset of a shared-memory pointer.  Flat shared addresses are
// {SHARED_BASE[31:0] in addr[63:32], lds_offset in addr[31:0]} (ISA 10.2),
// so truncation yields exactly the VDST operand GLOBAL_LOAD_ASYNC_TO_LDS
// expects.  Crucially this ptrtoint CAPTURES the shared array, so the
// "memory" clobbers on the asm below legally alias it and the fragment
// ds_loads cannot be constant-folded away.
static __device__ __forceinline__ unsigned lds_off(const void* p) {
  return (unsigned)(unsigned long long)p;
}

__global__ __launch_bounds__(256)
void gemm_bf16_wmma_kernel(const unsigned short* __restrict__ A,   // M x K bf16
                           const unsigned short* __restrict__ Bt,  // N x K bf16
                           float* __restrict__ Out,                // M x ldOut
                           const float* __restrict__ Res,          // optional
                           int N, int K, int ldOut, int addRes) {
  __shared__ unsigned short smem[2 * BUFHALVES];

  const int tid   = threadIdx.x;
  const int lane  = tid & 31;
  const int wave  = tid >> 5;
  const int wm    = wave & 3;      // 0..3 -> 32 rows each
  const int wn    = wave >> 2;     // 0..1 -> 64 cols each
  const int lhalf = lane & 15;
  const int hi    = lane >> 4;     // 0 or 1
  const int bm    = blockIdx.x * TILE_M;
  const int bn    = blockIdx.y * TILE_N;

  v8f acc[2][4];
  #pragma unroll
  for (int mt = 0; mt < 2; ++mt)
    #pragma unroll
    for (int nt = 0; nt < 4; ++nt)
      acc[mt][nt] = (v8f)(0.f);

  // async-stage one 128x32 A tile + 128x32 B tile into LDS buffer `buf`.
  // 4 x b128 per thread (2 for A, 2 for B). B row index is CLAMPED (not
  // branched): garbage columns only reach accumulators whose stores are
  // guarded by col < N, so no EXEC divergence on the load path.
  auto stage = [&](int k0, int buf) {
    const int base = buf * BUFHALVES;
    #pragma unroll
    for (int c = 0; c < 2; ++c) {
      const int chunk = tid + c * 256;       // 0..511
      const int row   = chunk >> 2;          // 0..127
      const int c8    = (chunk & 3) << 3;    // 0,8,16,24
      const unsigned long long ga =
          (unsigned long long)(A + (size_t)(bm + row) * K + k0 + c8);
      const unsigned la = lds_off(&smem[base + row * LDSROW + c8]);
      asm volatile("global_load_async_to_lds_b128 %0, %1, off"
                   :: "v"(la), "v"(ga) : "memory");
      int n = bn + row;
      n = (n < N) ? n : (N - 1);
      const unsigned long long gb =
          (unsigned long long)(Bt + (size_t)n * K + k0 + c8);
      const unsigned lb =
          lds_off(&smem[base + TILE_M * LDSROW + row * LDSROW + c8]);
      asm volatile("global_load_async_to_lds_b128 %0, %1, off"
                   :: "v"(lb), "v"(gb) : "memory");
    }
  };

  // prologue: fill buffer 0
  stage(0, 0);
  asm volatile("s_wait_asynccnt 0" ::: "memory");
  __syncthreads();

  int cur = 0;
  for (int k0 = 0; k0 < K; k0 += TILE_K) {
    // kick off async loads for the next K tile into the other buffer
    if (k0 + TILE_K < K) stage(k0 + TILE_K, cur ^ 1);

    const unsigned short* As = smem + cur * BUFHALVES;
    const unsigned short* Bs = As + TILE_M * LDSROW;

    // A fragments (ISA 7.12.2, 16-bit A 16x32):
    //   lanes 0-15: elems 0..7 = K 0..7,  elems 8..15 = K 16..23
    //   lanes16-31: elems 0..7 = K 8..15, elems 8..15 = K 24..31
    v16bf afrag[2];
    const int aoff = hi ? 8 : 0;
    #pragma unroll
    for (int mt = 0; mt < 2; ++mt) {
      const unsigned short* base = As + (wm * 32 + mt * 16 + lhalf) * LDSROW;
      afrag[mt] = cat8(*(const v8bf*)(base + aoff), *(const v8bf*)(base + aoff + 16));
    }
    // B fragments (16-bit B 32x16, N-major in LDS, K contiguous):
    //   lanes 0-15: K 0..7,16..23 ; lanes 16-31: K 8..15,24..31  (column = lane%16)
    v16bf bfrag[4];
    const int boff = hi ? 8 : 0;
    #pragma unroll
    for (int nt = 0; nt < 4; ++nt) {
      const unsigned short* base = Bs + (wn * 64 + nt * 16 + lhalf) * LDSROW;
      bfrag[nt] = cat8(*(const v8bf*)(base + boff), *(const v8bf*)(base + boff + 16));
    }

    #pragma unroll
    for (int mt = 0; mt < 2; ++mt)
      #pragma unroll
      for (int nt = 0; nt < 4; ++nt)
        acc[mt][nt] = __builtin_amdgcn_wmma_f32_16x16x32_bf16(
            false, afrag[mt], false, bfrag[nt], (short)0, acc[mt][nt], false, false);

    // wait for our own async loads of the next tile, then one workgroup
    // barrier: all LDS reads of `cur` finished (dscnt-waited before WMMA
    // consumption above), all async writes of `cur^1` are visible.
    asm volatile("s_wait_asynccnt 0" ::: "memory");
    __syncthreads();
    cur ^= 1;
  }

  // store D: VGPR i -> M = i + 8*hi (within tile), N = lane%16
  #pragma unroll
  for (int mt = 0; mt < 2; ++mt) {
    #pragma unroll
    for (int nt = 0; nt < 4; ++nt) {
      const int col = bn + wn * 64 + nt * 16 + lhalf;
      if (col < N) {
        const int row0 = bm + wm * 32 + mt * 16 + hi * 8;
        #pragma unroll
        for (int i = 0; i < 8; ++i) {
          const size_t idx = (size_t)(row0 + i) * ldOut + col;
          float v = acc[mt][nt][i];
          if (addRes) v += Res[idx];
          Out[idx] = v;
        }
      }
    }
  }
}

// ---------------- K3: depthwise causal conv (4 taps) + bias + SiLU ----------
__global__ __launch_bounds__(256)
void conv_silu_kernel(const float* __restrict__ zx, const float* __restrict__ cw,
                      const float* __restrict__ cb, float* __restrict__ convout) {
  const int c = blockIdx.x * 256 + threadIdx.x;     // 0..CONVDIM-1
  if (c >= CONVDIM) return;
  const int t = blockIdx.y;
  const int b = blockIdx.z;
  float acc = cb[c];
  #pragma unroll
  for (int j = 0; j < DCONV; ++j) {
    const int tt = t - (DCONV - 1) + j;
    if (tt >= 0)
      acc += cw[c * DCONV + j] * zx[((size_t)b * T_ + tt) * DINPROJ + DINNER + c];
  }
  convout[((size_t)b * T_ + t) * CONVDIM + c] = acc * sigmoidf_(acc);
}

// ---------------- K4: dt = softplus(dt_raw + dt_bias) ----------------
__global__ __launch_bounds__(256)
void dt_kernel(const float* __restrict__ zx, const float* __restrict__ dt_bias,
               float* __restrict__ dts) {
  const int i = blockIdx.x * 256 + threadIdx.x;     // ROWS*NHEADS
  if (i >= ROWS * NHEADS) return;
  const int row = i >> 5, h = i & 31;
  const float v = zx[(size_t)row * DINPROJ + (DINPROJ - NHEADS) + h] + dt_bias[h];
  dts[i] = (v > 20.f) ? v : logf(1.f + __expf(v));
}

// ---------------- K5: selective scan. 1 block = (batch, head) --------------
// 256 threads: p = tid&63 (HEADDIM), ng = tid>>6 (4 groups of 32 states).
// State h[p, n] lives in registers (32 f32 per lane).
__global__ __launch_bounds__(256)
void scan_kernel(const float* __restrict__ conv, const float* __restrict__ dts,
                 const float* __restrict__ A_log, const float* __restrict__ Dvec,
                 float* __restrict__ ys) {
  const int bh = blockIdx.x;          // 0..63
  const int b  = bh >> 5;
  const int h  = bh & 31;
  const int tid = threadIdx.x;
  const int p   = tid & 63;
  const int ng  = tid >> 6;
  const float Aneg = -__expf(A_log[h]);
  const float Dh   = Dvec[h];

  __shared__ float sBC[2 * DSTATE];   // [0..127]=B_t, [128..255]=C_t
  __shared__ float sx[HEADDIM];
  __shared__ float sred[256];
  __shared__ float sdt;

  float hst[32];
  #pragma unroll
  for (int j = 0; j < 32; ++j) hst[j] = 0.f;

  for (int t = 0; t < T_; ++t) {
    const size_t rowc = ((size_t)b * T_ + t) * CONVDIM;
    sBC[tid] = conv[rowc + DINNER + tid];
    if (tid < HEADDIM) sx[tid] = conv[rowc + (size_t)h * HEADDIM + tid];
    if (tid == 0) sdt = dts[((size_t)b * T_ + t) * NHEADS + h];
    __syncthreads();

    const float dt  = sdt;
    const float da  = __expf(dt * Aneg);
    const float xp  = sx[p];
    const float dtx = dt * xp;
    float partial = 0.f;
    #pragma unroll
    for (int j = 0; j < 32; ++j) {
      const int n = ng * 32 + j;
      const float hn = hst[j] * da + dtx * sBC[n];
      hst[j] = hn;
      partial += hn * sBC[DSTATE + n];
    }
    sred[tid] = partial;
    __syncthreads();
    if (ng == 0) {
      const float y = sred[p] + sred[p + 64] + sred[p + 128] + sred[p + 192] + Dh * xp;
      ys[((size_t)b * T_ + t) * DINNER + (size_t)h * HEADDIM + p] = y;
    }
    __syncthreads();
  }
}

// ---------------- K6: y = rmsnorm(y * silu(z)) -> bf16 ----------------
__global__ __launch_bounds__(256)
void gate_norm_pack_kernel(const float* __restrict__ ys, const float* __restrict__ zx,
                           const float* __restrict__ gw, unsigned short* __restrict__ yb) {
  __shared__ float s8[8];
  const int row = blockIdx.x;
  const int tid = threadIdx.x;
  const float* yr = ys + (size_t)row * DINNER;
  const float* zr = zx + (size_t)row * DINPROJ;   // z = first DINNER cols
  float v[8];
  float ss = 0.f;
  #pragma unroll
  for (int j = 0; j < 8; ++j) {
    const int i = tid + j * 256;
    const float z = zr[i];
    const float g = yr[i] * (z * sigmoidf_(z));
    v[j] = g;
    ss += g * g;
  }
  const float tot = block_sum_256(ss, s8);
  const float scale = rsqrtf(tot / (float)DINNER + 1e-5f);
  #pragma unroll
  for (int j = 0; j < 8; ++j) {
    const int i = tid + j * 256;
    yb[(size_t)row * DINNER + i] = f32_to_bf16(v[j] * scale * gw[i]);
  }
}

// ---------------- launch ----------------
extern "C" void kernel_launch(void* const* d_in, const int* in_sizes, int n_in,
                              void* d_out, int out_size, void* d_ws, size_t ws_size,
                              hipStream_t stream) {
  const float* x       = (const float*)d_in[0];
  const float* W_in    = (const float*)d_in[1];
  const float* conv_w  = (const float*)d_in[2];
  const float* conv_b  = (const float*)d_in[3];
  const float* dt_bias = (const float*)d_in[4];
  const float* A_log   = (const float*)d_in[5];
  const float* Dv      = (const float*)d_in[6];
  const float* norm_w  = (const float*)d_in[7];
  const float* gnorm_w = (const float*)d_in[8];
  const float* W_out   = (const float*)d_in[9];
  float* out = (float*)d_out;

  // workspace layout (bytes); total ~182 MB
  constexpr size_t OFF_XN   = 0;                                          // bf16 4096x1024
  constexpr size_t OFF_WIT  = OFF_XN  + (size_t)ROWS * DMODEL * 2;        // bf16 4384x1024
  constexpr size_t OFF_ZX   = OFF_WIT + (size_t)DINPROJ * DMODEL * 2;     // f32  4096x4384
  constexpr size_t OFF_CONV = OFF_ZX  + (size_t)ROWS * DINPROJ * 4;       // f32  4096x2304
  constexpr size_t OFF_DTS  = OFF_CONV+ (size_t)ROWS * CONVDIM * 4;       // f32  4096x32
  constexpr size_t OFF_YS   = OFF_DTS + (size_t)ROWS * NHEADS * 4;        // f32  4096x2048
  constexpr size_t OFF_YB   = OFF_YS  + (size_t)ROWS * DINNER * 4;        // bf16 4096x2048
  constexpr size_t OFF_WOT  = OFF_YB  + (size_t)ROWS * DINNER * 2;        // bf16 1024x2048

  char* ws = (char*)d_ws;
  unsigned short* xn   = (unsigned short*)(ws + OFF_XN);
  unsigned short* WiT  = (unsigned short*)(ws + OFF_WIT);
  float*          zx   = (float*)(ws + OFF_ZX);
  float*          conv = (float*)(ws + OFF_CONV);
  float*          dts  = (float*)(ws + OFF_DTS);
  float*          ys   = (float*)(ws + OFF_YS);
  unsigned short* yb   = (unsigned short*)(ws + OFF_YB);
  unsigned short* WoT  = (unsigned short*)(ws + OFF_WOT);

  // 0) rmsnorm + bf16 pack of activations
  rmsnorm_pack_kernel<<<ROWS, 256, 0, stream>>>(x, norm_w, xn);
  // 1) W_in^T bf16 pack (N=4384, K=1024)
  transpose_pack_kernel<<<(DINPROJ * DMODEL + 255) / 256, 256, 0, stream>>>(
      W_in, WiT, DMODEL, DINPROJ);
  // 2) GEMM1: zx = xn @ W_in   (M=4096, N=4384, K=1024)
  {
    dim3 grid(ROWS / TILE_M, (DINPROJ + TILE_N - 1) / TILE_N);
    gemm_bf16_wmma_kernel<<<grid, 256, 0, stream>>>(xn, WiT, zx, nullptr,
                                                    DINPROJ, DMODEL, DINPROJ, 0);
  }
  // 3) depthwise conv + SiLU
  {
    dim3 grid((CONVDIM + 255) / 256, T_, B_);
    conv_silu_kernel<<<grid, 256, 0, stream>>>(zx, conv_w, conv_b, conv);
  }
  // 4) dt softplus
  dt_kernel<<<(ROWS * NHEADS + 255) / 256, 256, 0, stream>>>(zx, dt_bias, dts);
  // 5) selective scan
  scan_kernel<<<B_ * NHEADS, 256, 0, stream>>>(conv, dts, A_log, Dv, ys);
  // 6) gate + rmsnorm + bf16 pack
  gate_norm_pack_kernel<<<ROWS, 256, 0, stream>>>(ys, zx, gnorm_w, yb);
  // 7) W_out^T bf16 pack (N=1024, K=2048)
  transpose_pack_kernel<<<(DMODEL * DINNER + 255) / 256, 256, 0, stream>>>(
      W_out, WoT, DINNER, DMODEL);
  // 8) GEMM2: out = yb @ W_out + x   (M=4096, N=1024, K=2048)
  {
    dim3 grid(ROWS / TILE_M, DMODEL / TILE_N);
    gemm_bf16_wmma_kernel<<<grid, 256, 0, stream>>>(yb, WoT, out, x,
                                                    DMODEL, DINNER, DMODEL, 1);
  }
}
